// StaticNeuralGraph_65747359367619
// MI455X (gfx1250) — matrix operand order repaired and checked by reading, not compiled, not run on hardware
//
#include <hip/hip_runtime.h>
#include <hip/hip_bf16.h>

typedef __attribute__((ext_vector_type(16))) _Float16 v16h;
typedef __attribute__((ext_vector_type(8)))  _Float16 v8h;
typedef __attribute__((ext_vector_type(8)))  float    v8f;

#define NTOT  100352      // B*H*W = 32*56*56
#define HW    3136
#define DIMC  800
#define EPS_  1e-5f

// ---------------- gw f32 -> f16 ----------------
__global__ void k_cvt_gw(const float* __restrict__ gw, _Float16* __restrict__ gwh) {
  int i = blockIdx.x * 256 + threadIdx.x;
  if (i < DIMC * DIMC) gwh[i] = (_Float16)gw[i];
}

// ---------------- stem: conv 3->16 s2 p1 + bias + BN + ReLU ----------------
__global__ void k_conv1(const float* __restrict__ x, const float* __restrict__ w,
                        const float* __restrict__ bias, const float* __restrict__ bg,
                        const float* __restrict__ bb, const float* __restrict__ bm,
                        const float* __restrict__ bv, float* __restrict__ out) {
  int idx = blockIdx.x * 256 + threadIdx.x;         // 32*16*112*112
  int ox = idx % 112; int t = idx / 112;
  int oy = t % 112;   t /= 112;
  int c  = t % 16;    int b = t / 16;
  float acc = bias[c];
  for (int ci = 0; ci < 3; ++ci)
    for (int ky = 0; ky < 3; ++ky) {
      int iy = 2 * oy - 1 + ky;
      if (iy < 0 || iy >= 224) continue;
      for (int kx = 0; kx < 3; ++kx) {
        int ix = 2 * ox - 1 + kx;
        if (ix < 0 || ix >= 224) continue;
        acc += x[(((size_t)b * 3 + ci) * 224 + iy) * 224 + ix] *
               w[((c * 3 + ci) * 3 + ky) * 3 + kx];
      }
    }
  float sc = bg[c] * rsqrtf(bv[c] + EPS_);
  acc = (acc - bm[c]) * sc + bb[c];
  out[idx] = acc > 0.f ? acc : 0.f;
}

// ---------------- stem: depthwise 3x3 s2 p1 + bias ----------------
__global__ void k_dw1(const float* __restrict__ in, const float* __restrict__ w,
                      const float* __restrict__ bias, float* __restrict__ out) {
  int idx = blockIdx.x * 256 + threadIdx.x;         // 32*16*56*56
  int ox = idx % 56; int t = idx / 56;
  int oy = t % 56;   t /= 56;
  int c  = t % 16;   int b = t / 16;
  float acc = bias[c];
  for (int ky = 0; ky < 3; ++ky) {
    int iy = 2 * oy - 1 + ky;
    if (iy < 0 || iy >= 112) continue;
    for (int kx = 0; kx < 3; ++kx) {
      int ix = 2 * ox - 1 + kx;
      if (ix < 0 || ix >= 112) continue;
      acc += in[(((size_t)b * 16 + c) * 112 + iy) * 112 + ix] * w[c * 9 + ky * 3 + kx];
    }
  }
  out[idx] = acc;
}

// ------- stem: 1x1 conv 16->32 + bias + BN, writes channel-major H[c][n]; zeros c>=32 -------
__global__ void k_pw_bn(const float* __restrict__ d1, const float* __restrict__ w2,
                        const float* __restrict__ b2, const float* __restrict__ bg,
                        const float* __restrict__ bb, const float* __restrict__ bm,
                        const float* __restrict__ bv, float* __restrict__ H) {
  size_t idx = (size_t)blockIdx.x * 256 + threadIdx.x;   // 800 * 100352
  int n = (int)(idx % NTOT);
  int c = (int)(idx / NTOT);
  float val = 0.f;
  if (c < 32) {
    int b = n / HW, hw = n % HW;
    float acc = b2[c];
    for (int k = 0; k < 16; ++k)
      acc += d1[((size_t)b * 16 + k) * HW + hw] * w2[c * 16 + k];
    float sc = bg[c] * rsqrtf(bv[c] + EPS_);
    val = (acc - bm[c]) * sc + bb[c];
  }
  H[idx] = val;
}

// ---------------- per (c,b) instance-norm stats ----------------
__global__ void k_stats(const float* __restrict__ H, float* __restrict__ mean,
                        float* __restrict__ rstd, int r0) {
  int c = blockIdx.x >> 5;
  int b = blockIdx.x & 31;
  const float* p = H + (size_t)(r0 + c) * NTOT + (size_t)b * HW;
  float s = 0.f, sq = 0.f;
  for (int i = threadIdx.x; i < HW; i += 256) { float v = p[i]; s += v; sq += v * v; }
  __shared__ float ss[256], s2[256];
  ss[threadIdx.x] = s; s2[threadIdx.x] = sq; __syncthreads();
  for (int off = 128; off > 0; off >>= 1) {
    if (threadIdx.x < off) {
      ss[threadIdx.x] += ss[threadIdx.x + off];
      s2[threadIdx.x] += s2[threadIdx.x + off];
    }
    __syncthreads();
  }
  if (threadIdx.x == 0) {
    float mu  = ss[0] * (1.f / HW);
    float var = s2[0] * (1.f / HW) - mu * mu;
    mean[c * 32 + b] = mu;
    rstd[c * 32 + b] = rsqrtf(var + EPS_);
  }
}

// ------- norm + affine + ReLU + depthwise 3x3 p1 -> f16 xaT[n][k] (K-contiguous for WMMA B) -------
__global__ void k_norm_dw(const float* __restrict__ H, const float* __restrict__ mean,
                          const float* __restrict__ rstd, const float* __restrict__ gamma,
                          const float* __restrict__ beta, const float* __restrict__ cdw,
                          _Float16* __restrict__ xaT, int r0, int s) {
  size_t idx = (size_t)blockIdx.x * 256 + threadIdx.x;   // s * NTOT
  if (idx >= (size_t)s * NTOT) return;
  int c = (int)(idx % s);
  int n = (int)(idx / s);
  int b = n / HW, hw = n % HW;
  int y = hw / 56, x = hw % 56;
  int gc = r0 + c;
  float mu = mean[c * 32 + b], rs = rstd[c * 32 + b];
  float ga = gamma[gc] * rs;
  float be = beta[gc] - mu * ga;
  const float* hp = H + (size_t)gc * NTOT + (size_t)b * HW;
  float acc = 0.f;
  for (int ky = 0; ky < 3; ++ky) {
    int iy = y - 1 + ky; if (iy < 0 || iy >= 56) continue;
    for (int kx = 0; kx < 3; ++kx) {
      int ix = x - 1 + kx; if (ix < 0 || ix >= 56) continue;
      float v = hp[iy * 56 + ix] * ga + be;
      v = v > 0.f ? v : 0.f;
      acc += v * cdw[gc * 9 + ky * 3 + kx];
    }
  }
  xaT[(size_t)n * s + c] = (_Float16)acc;
}

// ------- WMMA GEMM: H[row0+m][n] += sum_k gw16[row0+m][r0+k] * xaT[n][k] -------
// one wave -> 16(M) x 64(N) strip, K stepped by 32, v_wmma_f32_16x16x32_f16
__global__ void __launch_bounds__(256)
k_gemm(const _Float16* __restrict__ gwh, const _Float16* __restrict__ xaT,
       float* __restrict__ H, int mTiles, int r0, int row0, int kBlocks, int s) {
  int lane = threadIdx.x & 31;
  int wIn  = threadIdx.x >> 5;
  long wid = (long)blockIdx.x * 8 + wIn;
  int mTile = (int)(wid % mTiles);
  int nBase = (int)(wid / mTiles) * 64;
  int l16 = lane & 15, g = lane >> 4;

  // A: lane holds gw row (row0+mTile*16+l16); K contiguous, +16 halves for upper lane half
  const _Float16* Arow = gwh + (size_t)(row0 + mTile * 16 + l16) * DIMC + r0;
  // B: lane holds column n = nBase + q*16 + l16; K contiguous, +16 halves for upper lane half
  const _Float16* Bb = xaT + (size_t)(nBase + l16) * s + 16 * g;

  v8f a0 = {}, a1 = {}, a2 = {}, a3 = {};
  for (int kb = 0; kb < kBlocks; ++kb) {
    const _Float16* ap = Arow + kb * 32;
    v8h alo = *(const v8h*)(ap + 16 * g);       // K = kbase + 16g .. +7   (elems 0..7)
    v8h ahi = *(const v8h*)(ap + 16 * g + 8);   // K = kbase+16g+8 .. +7  (elems 8..15)
    v16h A;
#pragma unroll
    for (int e = 0; e < 8; ++e) { A[e] = alo[e]; A[8 + e] = ahi[e]; }
    v16h B0 = *(const v16h*)(Bb + (size_t)0  * s + kb * 32);
    v16h B1 = *(const v16h*)(Bb + (size_t)16 * s + kb * 32);
    v16h B2 = *(const v16h*)(Bb + (size_t)32 * s + kb * 32);
    v16h B3 = *(const v16h*)(Bb + (size_t)48 * s + kb * 32);
    a0 = __builtin_amdgcn_wmma_f32_16x16x32_f16(false, A, false, B0, (short)0, a0, false, false);
    a1 = __builtin_amdgcn_wmma_f32_16x16x32_f16(false, A, false, B1, (short)0, a1, false, false);
    a2 = __builtin_amdgcn_wmma_f32_16x16x32_f16(false, A, false, B2, (short)0, a2, false, false);
    a3 = __builtin_amdgcn_wmma_f32_16x16x32_f16(false, A, false, B3, (short)0, a3, false, false);
  }
  // C/D layout: VGPR r, lane -> (M = r + 8g, N = l16)
  float* Hp = H + (size_t)(row0 + mTile * 16 + 8 * g) * NTOT + nBase + l16;
#pragma unroll
  for (int r = 0; r < 8; ++r) {
    float* hp = Hp + (size_t)r * NTOT;
    hp[0]  += a0[r];
    hp[16] += a1[r];
    hp[32] += a2[r];
    hp[48] += a3[r];
  }
}

// ---------------- head ----------------
__global__ void k_feat(const float* __restrict__ H, float* __restrict__ feat) {
  int i = blockIdx.x * 256 + threadIdx.x;           // 32*192
  if (i >= 32 * 192) return;
  int f = i % 192, b = i / 192;
  const float* p = H + (size_t)(608 + f) * NTOT + (size_t)b * HW;
  feat[b * 192 + f] = 0.25f * (p[26 * 56 + 26] + p[26 * 56 + 27] +
                               p[27 * 56 + 26] + p[27 * 56 + 27]);
}

__global__ void k_fc(const float* __restrict__ feat, const float* __restrict__ fw,
                     const float* __restrict__ fb, float* __restrict__ out) {
  int i = blockIdx.x * 256 + threadIdx.x;           // 32*1000
  if (i >= 32 * 1000) return;
  int o = i % 1000, b = i / 1000;
  const float* fv = feat + b * 192;
  const float* wr = fw + (size_t)o * 192;
  float acc = fb[o];
  for (int k = 0; k < 192; ++k) acc += fv[k] * wr[k];
  out[i] = acc;
}

extern "C" void kernel_launch(void* const* d_in, const int* in_sizes, int n_in,
                              void* d_out, int out_size, void* d_ws, size_t ws_size,
                              hipStream_t stream) {
  const float* x        = (const float*)d_in[0];
  const float* ds_w1    = (const float*)d_in[1];
  const float* ds_b1    = (const float*)d_in[2];
  const float* bn1_g    = (const float*)d_in[3];
  const float* bn1_b    = (const float*)d_in[4];
  const float* bn1_m    = (const float*)d_in[5];
  const float* bn1_v    = (const float*)d_in[6];
  const float* ds_dw    = (const float*)d_in[7];
  const float* ds_dwb   = (const float*)d_in[8];
  const float* ds_w2    = (const float*)d_in[9];
  const float* ds_b2    = (const float*)d_in[10];
  const float* bn2_g    = (const float*)d_in[11];
  const float* bn2_b    = (const float*)d_in[12];
  const float* bn2_m    = (const float*)d_in[13];
  const float* bn2_v    = (const float*)d_in[14];
  const float* conv_dw  = (const float*)d_in[15];
  const float* gw       = (const float*)d_in[16];
  const float* in_gamma = (const float*)d_in[17];
  const float* in_beta  = (const float*)d_in[18];
  const float* fc_w     = (const float*)d_in[19];
  const float* fc_b     = (const float*)d_in[20];

  char* ws = (char*)d_ws;
  size_t off = 0;
  auto take = [&](size_t bytes) -> char* {
    char* p = ws + off;
    off += (bytes + 255) & ~(size_t)255;
    return p;
  };
  float*    H    = (float*)take((size_t)DIMC * NTOT * 4);        // 321.1 MB, channel-major state
  _Float16* xaT  = (_Float16*)take((size_t)192 * NTOT * 2);      // 38.5 MB, B-matrix staging
  float*    c1   = (float*)take((size_t)32 * 16 * 112 * 112 * 4);
  float*    dd1  = (float*)take((size_t)32 * 16 * HW * 4);
  _Float16* gwh  = (_Float16*)take((size_t)DIMC * DIMC * 2);
  float*    mean = (float*)take((size_t)192 * 32 * 4);
  float*    rstd = (float*)take((size_t)192 * 32 * 4);
  float*    feat = (float*)take((size_t)32 * 192 * 4);
  (void)ws_size; (void)in_sizes; (void)n_in; (void)out_size;

  k_cvt_gw<<<2500, 256, 0, stream>>>(gw, gwh);
  k_conv1 <<<25088, 256, 0, stream>>>(x, ds_w1, ds_b1, bn1_g, bn1_b, bn1_m, bn1_v, c1);
  k_dw1   <<<6272, 256, 0, stream>>>(c1, ds_dw, ds_dwb, dd1);
  k_pw_bn <<<313600, 256, 0, stream>>>(dd1, ds_w2, ds_b2, bn2_g, bn2_b, bn2_m, bn2_v, H);

  const int r0s[5]   = {0, 32, 224, 416, 608};
  const int ss[5]    = {32, 192, 192, 192, 192};
  const int row0s[5] = {32, 224, 416, 608, 608};
  for (int i = 0; i < 5; ++i) {
    int r0 = r0s[i], s = ss[i], row0 = row0s[i];
    k_stats<<<s * 32, 256, 0, stream>>>(H, mean, rstd, r0);
    size_t tot = (size_t)s * NTOT;
    k_norm_dw<<<(unsigned)((tot + 255) / 256), 256, 0, stream>>>(
        H, mean, rstd, in_gamma, in_beta, conv_dw, xaT, r0, s);
    int M = DIMC - row0, mTiles = M / 16, kBlocks = s / 32;
    unsigned blocks = (unsigned)((size_t)mTiles * (NTOT / 64) / 8);  // waves/8, exact
    k_gemm<<<blocks, 256, 0, stream>>>(gwh, xaT, H, mTiles, r0, row0, kBlocks, s);
  }
  k_feat<<<24, 256, 0, stream>>>(H, feat);
  k_fc  <<<125, 256, 0, stream>>>(feat, fc_w, fc_b, (float*)d_out);
}
